// GNN_34248069218823
// MI455X (gfx1250) — compile-verified, matrix operand order checked by reading, not run on hardware
//
#include <hip/hip_runtime.h>
#include <math.h>

#define NF 128   // conv feature width
#define H1 384
#define H2 256
#define LDW 136  // padded LDS row stride (halves): bank-conflict-free column reads

typedef __attribute__((ext_vector_type(16))) _Float16 v16h;
typedef __attribute__((ext_vector_type(8)))  _Float16 v8h;
typedef __attribute__((ext_vector_type(4)))  _Float16 v4h;
typedef __attribute__((ext_vector_type(8)))  float    v8f;
typedef __attribute__((ext_vector_type(4)))  int      v4i;
typedef __attribute__((address_space(1))) v4i* gv4i_p;  // global int4*
typedef __attribute__((address_space(3))) v4i* lv4i_p;  // LDS int4*

#ifndef __has_builtin
#define __has_builtin(x) 0
#endif
#if __has_builtin(__builtin_amdgcn_global_load_async_to_lds_b128)
#define USE_ASYNC_LDS 1
#else
#define USE_ASYNC_LDS 0
#endif

__device__ __forceinline__ float eluf(float x) { return x > 0.f ? x : expm1f(x); }

// order-preserving float <-> uint encoding for atomic max
__device__ __forceinline__ unsigned f2ord(float x) {
  unsigned u = __float_as_uint(x);
  return (u & 0x80000000u) ? ~u : (u | 0x80000000u);
}
__device__ __forceinline__ float ord2f(unsigned k) {
  unsigned u = (k & 0x80000000u) ? (k & 0x7FFFFFFFu) : ~k;
  return __uint_as_float(u);
}

// ---------------- degree / CSR build (once, reused by all 3 layers) ----------------
__global__ void k_zero_i32(int* p, int n) {
  int i = blockIdx.x * blockDim.x + threadIdx.x;
  if (i < n) p[i] = 0;
}
__global__ void k_degE(const int* __restrict__ dst, int* degE, int e) {
  int i = blockIdx.x * blockDim.x + threadIdx.x;
  if (i < e) atomicAdd(&degE[dst[i]], 1);
}
__global__ void k_dinv(const int* __restrict__ degE, float* dinv, int n) {
  int i = blockIdx.x * blockDim.x + threadIdx.x;
  if (i < n) dinv[i] = rsqrtf((float)degE[i] + 1.0f);  // +1 self-loop
}
// exclusive scan, 3-phase (N <= 256*512)
__global__ void k_scan1(const int* __restrict__ in, int* out, int* blockSums, int n) {
  __shared__ int sh[256];
  int gid = blockIdx.x * 256 + threadIdx.x;
  int v = (gid < n) ? in[gid] : 0;
  sh[threadIdx.x] = v;
  __syncthreads();
  for (int s = 1; s < 256; s <<= 1) {
    int t = (threadIdx.x >= s) ? sh[threadIdx.x - s] : 0;
    __syncthreads();
    sh[threadIdx.x] += t;
    __syncthreads();
  }
  if (gid < n) out[gid] = sh[threadIdx.x] - v;        // exclusive
  if (threadIdx.x == 255) blockSums[blockIdx.x] = sh[255];
}
__global__ void k_scan2(int* blockSums, int nb) {      // single block, nb <= 512
  __shared__ int sh[512];
  int v = ((int)threadIdx.x < nb) ? blockSums[threadIdx.x] : 0;
  sh[threadIdx.x] = v;
  __syncthreads();
  for (int s = 1; s < 512; s <<= 1) {
    int t = (threadIdx.x >= s) ? sh[threadIdx.x - s] : 0;
    __syncthreads();
    sh[threadIdx.x] += t;
    __syncthreads();
  }
  if ((int)threadIdx.x < nb) blockSums[threadIdx.x] = sh[threadIdx.x] - v;
}
__global__ void k_scan3(int* out, const int* __restrict__ blockSums, int n) {
  int gid = blockIdx.x * 256 + threadIdx.x;
  if (gid < n) out[gid] += blockSums[blockIdx.x];
}
__global__ void k_copy_i32(const int* __restrict__ a, int* b, int n) {
  int i = blockIdx.x * blockDim.x + threadIdx.x;
  if (i < n) b[i] = a[i];
}
__global__ void k_bin(const int* __restrict__ src, const int* __restrict__ dst,
                      int* cursor, int* sortedSrc, int e) {
  int i = blockIdx.x * blockDim.x + threadIdx.x;
  if (i >= e) return;
  int pos = atomicAdd(&cursor[dst[i]], 1);
  sortedSrc[pos] = src[i];
}

// Convert W (f32 row-major [k][c]) -> f16 transposed [c][k], once per layer.
__global__ void k_w2h(const float* __restrict__ Wf, _Float16* __restrict__ WhT) {
  int idx = blockIdx.x * blockDim.x + threadIdx.x;
  if (idx >= NF * NF) return;
  int k = idx >> 7, c = idx & 127;
  WhT[c * NF + k] = (_Float16)Wf[idx];
}

// ---------------- WMMA GEMM: mscaled[M][c] = dinv[M] * (A @ W)[M][c] (f16) ----------------
// One block = 8 waves = 128 rows. WhT is pre-transposed f16 [col][k].
__global__ void __launch_bounds__(256)
k_gcn_gemm(const float* __restrict__ A, int lda, const _Float16* __restrict__ WhT,
           const float* __restrict__ dinv, _Float16* __restrict__ mscaled, int nrows) {
  __shared__ _Float16 Wl[NF * LDW];  // [col][k], padded rows

#if USE_ASYNC_LDS
  {
    // 2048 x 16B chunks, 8 per thread: direct global->LDS async copy (ASYNCcnt).
    const char* g = (const char*)WhT;
    char* l = (char*)Wl;
    #pragma unroll
    for (int i = 0; i < 8; ++i) {
      int ch = threadIdx.x + i * 256;
      int col = ch >> 4, wi = ch & 15;
      __builtin_amdgcn_global_load_async_to_lds_b128(
          (gv4i_p)(g + ch * 16), (lv4i_p)(l + col * (LDW * 2) + wi * 16), 0, 0);
    }
    asm volatile("s_wait_asynccnt 0x0" ::: "memory");
  }
#else
  for (int ch = threadIdx.x; ch < 2048; ch += 256) {
    int col = ch >> 4, wi = ch & 15;
    *(v8h*)((char*)Wl + col * (LDW * 2) + wi * 16) = ((const v8h*)WhT)[ch];
  }
#endif
  __syncthreads();

  int t = threadIdx.x;
  int wave = t >> 5, lane = t & 31;
  int r0 = blockIdx.x * 128 + wave * 16;
  if (r0 >= nrows) return;                 // uniform per wave; EXEC all-1s below

  int half = lane >> 4;                    // 0: lanes 0-15, 1: lanes 16-31
  int lm   = lane & 15;
  int off  = half * 8;                     // ISA 16-bit A/B lane split: K 0..7 vs 8..15
  int Mrow = r0 + lm;
  if (Mrow >= nrows) Mrow = nrows - 1;     // clamp, store-side guarded
  const float* rowp = A + (size_t)Mrow * lda;

  // Prefetch + convert the full A row fragment set up front (hide global latency).
  v16h afrag[4];
  #pragma unroll
  for (int ki = 0; ki < 4; ++ki) {
    int kb = ki * 32;
    float4 p0 = *(const float4*)(rowp + kb + off);
    float4 p1 = *(const float4*)(rowp + kb + off + 4);
    float4 p2 = *(const float4*)(rowp + kb + 16 + off);
    float4 p3 = *(const float4*)(rowp + kb + 16 + off + 4);
    v16h a;
    a[0]=(_Float16)p0.x; a[1]=(_Float16)p0.y; a[2]=(_Float16)p0.z; a[3]=(_Float16)p0.w;
    a[4]=(_Float16)p1.x; a[5]=(_Float16)p1.y; a[6]=(_Float16)p1.z; a[7]=(_Float16)p1.w;
    a[8]=(_Float16)p2.x; a[9]=(_Float16)p2.y; a[10]=(_Float16)p2.z; a[11]=(_Float16)p2.w;
    a[12]=(_Float16)p3.x; a[13]=(_Float16)p3.y; a[14]=(_Float16)p3.z; a[15]=(_Float16)p3.w;
    afrag[ki] = a;
  }

  v8f acc[8] = {};                         // 8 column tiles of 16 -> 128 cols
  #pragma unroll
  for (int ki = 0; ki < 4; ++ki) {
    int kb = ki * 32;
    #pragma unroll
    for (int tc = 0; tc < 8; ++tc) {
      const _Float16* colp = &Wl[(tc * 16 + lm) * LDW];
      v8h blo = *(const v8h*)(colp + kb + off);
      v8h bhi = *(const v8h*)(colp + kb + 16 + off);
      v16h b;
      #pragma unroll
      for (int q = 0; q < 8; ++q) { b[q] = blo[q]; b[8 + q] = bhi[q]; }
      acc[tc] = __builtin_amdgcn_wmma_f32_16x16x32_f16(
          false, afrag[ki], false, b, (short)0, acc[tc], false, false);
    }
  }

  // epilogue: scale rows by dinv, emit f16 gather copy
  float dv[8];
  #pragma unroll
  for (int j = 0; j < 8; ++j) {
    int M = r0 + j + half * 8;
    dv[j] = (M < nrows) ? dinv[M] : 0.f;
  }
  #pragma unroll
  for (int tc = 0; tc < 8; ++tc) {
    int col = tc * 16 + lm;
    #pragma unroll
    for (int j = 0; j < 8; ++j) {
      int M = r0 + j + half * 8;
      if (M < nrows)
        mscaled[(size_t)M * NF + col] = (_Float16)(dv[j] * acc[tc][j]);
    }
  }
}

// ---------------- CSR aggregate: one wave per node, zero atomics ----------------
// comb[d, layerOff+f] = elu(dinv[d] * (sum_{s in in(d)} mscaled[s,f] + mscaled[d,f]))
__global__ void __launch_bounds__(256)
k_aggregate(const int* __restrict__ offs, const int* __restrict__ degE,
            const int* __restrict__ sortedSrc, const _Float16* __restrict__ mscaled,
            const float* __restrict__ dinv, float* __restrict__ comb,
            int layerOff, int n) {
  int d = (blockIdx.x * 256 + threadIdx.x) >> 5;  // one wave32 per node
  if (d >= n) return;
  int lane = threadIdx.x & 31;
  int beg = offs[d], cnt = degE[d];
  float a0, a1, a2, a3;
  {  // self-loop term: dinv[d]*m[d]
    v4h v = *(const v4h*)(mscaled + (size_t)d * NF + lane * 4);
    a0 = (float)v[0]; a1 = (float)v[1]; a2 = (float)v[2]; a3 = (float)v[3];
  }
  for (int k = 0; k < cnt; ++k) {
    int s = sortedSrc[beg + k];               // uniform across wave -> 1 request
    v4h v = *(const v4h*)(mscaled + (size_t)s * NF + lane * 4);
    a0 += (float)v[0]; a1 += (float)v[1]; a2 += (float)v[2]; a3 += (float)v[3];
  }
  float dvn = dinv[d];
  float4 r;
  r.x = eluf(dvn * a0); r.y = eluf(dvn * a1);
  r.z = eluf(dvn * a2); r.w = eluf(dvn * a3);
  *(float4*)(comb + (size_t)d * H1 + layerOff + lane * 4) = r;
}

// ---------------- pooling ----------------
__global__ void k_pool_init(unsigned* pooledU, int g) {
  int idx = blockIdx.x * blockDim.x + threadIdx.x;
  if (idx < g * H1) pooledU[idx] = 0x007FFFFFu;  // f2ord(-inf)
}
__global__ void k_pool(const float* __restrict__ comb, const int* __restrict__ batch,
                       unsigned* pooledU, int n) {
  int idx = blockIdx.x * blockDim.x + threadIdx.x;
  if (idx >= n * H1) return;
  int i = idx / H1, f = idx % H1;
  atomicMax(&pooledU[(size_t)batch[i] * H1 + f], f2ord(comb[idx]));
}
__global__ void k_pool_fin(const unsigned* __restrict__ pooledU, float* pooledF, int g) {
  int idx = blockIdx.x * blockDim.x + threadIdx.x;
  if (idx < g * H1) pooledF[idx] = eluf(ord2f(pooledU[idx]));
}

// ---------------- MLP head ----------------
__global__ void k_mlp1(const float* __restrict__ pooledF, const float* __restrict__ L1w,
                       const float* __restrict__ L1b, float* hid, int g) {
  int idx = blockIdx.x * blockDim.x + threadIdx.x;
  if (idx >= g * H2) return;
  int gi = idx >> 8, j = idx & 255;
  float s = L1b[j];
  const float* pr = pooledF + (size_t)gi * H1;
  for (int k = 0; k < H1; ++k) s += pr[k] * L1w[k * H2 + j];
  hid[idx] = eluf(s);
}
__global__ void k_mlp2(const float* __restrict__ hid, const float* __restrict__ L2w,
                       const float* __restrict__ L2b, float* out, int g) {
  int gi = blockIdx.x * blockDim.x + threadIdx.x;
  if (gi >= g) return;
  float s = L2b[0];
  const float* hr = hid + (size_t)gi * H2;
  for (int k = 0; k < H2; ++k) s += hr[k] * L2w[k];
  out[gi] = 1.f / (1.f + expf(-s));
}

extern "C" void kernel_launch(void* const* d_in, const int* in_sizes, int n_in,
                              void* d_out, int out_size, void* d_ws, size_t ws_size,
                              hipStream_t stream) {
  const float* x    = (const float*)d_in[0];
  const int*   ei   = (const int*)  d_in[1];
  const int*   batch= (const int*)  d_in[2];
  const float* W1   = (const float*)d_in[3];
  const float* W2   = (const float*)d_in[4];
  const float* W3   = (const float*)d_in[5];
  const float* L1w  = (const float*)d_in[6];
  const float* L1b  = (const float*)d_in[7];
  const float* L2w  = (const float*)d_in[8];
  const float* L2b  = (const float*)d_in[9];

  const int N = in_sizes[0] / NF;
  const int E = in_sizes[1] / 2;
  const int G = out_size;
  const int* src = ei;
  const int* dst = ei + E;

  char* ws = (char*)d_ws;
  size_t off = 0;
  auto alloc = [&](size_t bytes) -> void* {
    void* p = ws + off;
    off = (off + bytes + 255) & ~(size_t)255;
    return p;
  };
  int*       degE      = (int*)      alloc((size_t)N * 4);
  int*       offs      = (int*)      alloc((size_t)N * 4);
  int*       cursor    = (int*)      alloc((size_t)N * 4);
  int*       blockSums = (int*)      alloc((size_t)512 * 4);
  int*       sortedSrc = (int*)      alloc((size_t)E * 4);
  float*     dinv      = (float*)    alloc((size_t)N * 4);
  _Float16*  mscaled   = (_Float16*) alloc((size_t)N * NF * 2);
  float*     comb      = (float*)    alloc((size_t)N * H1 * 4);
  _Float16*  Wh        = (_Float16*) alloc((size_t)NF * NF * 2);
  unsigned*  pooledU   = (unsigned*) alloc((size_t)G * H1 * 4);
  float*     pooledF   = (float*)    alloc((size_t)G * H1 * 4);
  float*     hid       = (float*)    alloc((size_t)G * H2 * 4);
  (void)ws_size; (void)n_in;

  const int TB = 256;
  auto nb = [](int n, int b) { return (n + b - 1) / b; };

  // ---- build in-degree CSR once (reused by all three conv layers) ----
  int nScanBlocks = nb(N, 256);
  k_zero_i32<<<nb(N, TB), TB, 0, stream>>>(degE, N);
  k_degE    <<<nb(E, TB), TB, 0, stream>>>(dst, degE, E);
  k_dinv    <<<nb(N, TB), TB, 0, stream>>>(degE, dinv, N);
  k_scan1   <<<nScanBlocks, 256, 0, stream>>>(degE, offs, blockSums, N);
  k_scan2   <<<1, 512, 0, stream>>>(blockSums, nScanBlocks);
  k_scan3   <<<nScanBlocks, 256, 0, stream>>>(offs, blockSums, N);
  k_copy_i32<<<nb(N, TB), TB, 0, stream>>>(offs, cursor, N);
  k_bin     <<<nb(E, TB), TB, 0, stream>>>(src, dst, cursor, sortedSrc, E);

  auto run_layer = [&](const float* Ain, int lda, const float* Wf, int layerOff) {
    k_w2h      <<<nb(NF * NF, TB), TB, 0, stream>>>(Wf, Wh);
    k_gcn_gemm <<<nb(N, 128), 256, 0, stream>>>(Ain, lda, Wh, dinv, mscaled, N);
    k_aggregate<<<nb(N * 32, 256), 256, 0, stream>>>(offs, degE, sortedSrc, mscaled,
                                                     dinv, comb, layerOff, N);
  };
  run_layer(x,          NF, W1, 0);
  run_layer(comb,       H1, W2, 128);
  run_layer(comb + 128, H1, W3, 256);

  k_pool_init<<<nb(G * H1, TB), TB, 0, stream>>>(pooledU, G);
  k_pool     <<<nb(N * H1, TB), TB, 0, stream>>>(comb, batch, pooledU, N);
  k_pool_fin <<<nb(G * H1, TB), TB, 0, stream>>>(pooledU, pooledF, G);
  k_mlp1     <<<nb(G * H2, TB), TB, 0, stream>>>(pooledF, L1w, L1b, hid, G);
  k_mlp2     <<<nb(G, TB), TB, 0, stream>>>(hid, L2w, L2b, (float*)d_out, G);
}
